// DomainPriorInitFusionLayer_3384434230014
// MI455X (gfx1250) — matrix operand order repaired and checked by reading, not compiled, not run on hardware
//
#include <hip/hip_runtime.h>
#include <hip/hip_bf16.h>
#include <math.h>

// ---------------------------------------------------------------------------
// DomainPriorInitFusionLayer forward for MI455X (gfx1250, wave32, WMMA).
// All large GEMMs run through v_wmma_f32_16x16x32_f16 with f32 accumulation.
// Weights are pre-converted once per launch to f16, transposed [N][K], so the
// GEMM B-staging is a single 16B load + single ds_store_b128 per thread.
// A/B tiles are double-buffered in LDS (one barrier per K-chunk) so global
// loads for chunk c+1 overlap the WMMAs of chunk c.
// ---------------------------------------------------------------------------

typedef __attribute__((ext_vector_type(16))) _Float16 v16h;
typedef __attribute__((ext_vector_type(8)))  _Float16 v8h;
typedef __attribute__((ext_vector_type(4)))  _Float16 v4h;
typedef __attribute__((ext_vector_type(8)))  float    v8f;

#define BB   2
#define CC   256
#define LL   4096      // D*H*W = 16*16*16
#define LP   50
#define NH   8
#define DH   32

#define BM 128
#define BN 64
#define BK 32
#define ASTRIDE 48     // f16 elements; 96 bytes/row (multiple of 16B)

__device__ __forceinline__ float waveRedSum(float v) {
#pragma unroll
  for (int off = 16; off > 0; off >>= 1) v += __shfl_xor(v, off, 32);
  return v;
}
__device__ __forceinline__ float waveRedMax(float v) {
#pragma unroll
  for (int off = 16; off > 0; off >>= 1) v = fmaxf(v, __shfl_xor(v, off, 32));
  return v;
}
__device__ __forceinline__ float lrelu(float v) { return v >= 0.f ? v : 0.01f * v; }
__device__ __forceinline__ float gelu(float v)  { return 0.5f * v * (1.f + erff(v * 0.7071067811865476f)); }

// ---------------------------------------------------------------------------
// Generic WMMA GEMM:  out = epilogue(A @ B)   A: M x K (gathered), B: K x N.
// B is given pre-transposed/pre-converted: Bw16[n*K + k] (f16).
// A gather modes (per part; k < ksplit -> part0, else part1 with k-=ksplit):
//   0: plain row-major:             A[m*lda + k]            (k-stride 1)
//   1: channel-first (m = b*L+s):   A[(b*ch + k)*L + s]     (k-stride L)
//   2: channel-first, m=(b*L+s)*6+j, j ignored              (k-stride L)
//   3: edge layout,   m=(b*L+s)*6+j: A[((b*ch+k)*L+s)*6+j]  (k-stride 6L)
// Epilogue: v = gemm; +bias[n]; BN over n (g,b,m,v); act(0 none,1 gelu,2 lrelu);
//           +res (mode 0 plain / 1 channel-first); store (mode 0 plain / 1 CF).
// Requirements: K % 32 == 0, N % 64 == 0, ksplit % 32 == 0.
// ---------------------------------------------------------------------------
struct GemmP {
  const float* A0; const float* A1;
  const _Float16* Bw;            // transposed f16 weights [N][K]
  const float* bias;
  const float* bng; const float* bnb; const float* bnm; const float* bnv;
  const float* res;
  float* out;
  int M, N, K;
  int ksplit;
  int amode0, amode1;
  int lda0, lda1;
  int ch0, ch1;
  int L;
  int act;
  int resmode;
  int outmode;
};

__device__ __forceinline__ const float* a_base(const float* A, int mode, int m,
                                               int lda, int ch, int L) {
  switch (mode) {
    default:
    case 0: return A + m * lda;
    case 1: { int b = m / L, s = m - b * L; return A + (b * ch) * L + s; }
    case 2: { int t = m / 6; int b = t / L, s = t - b * L; return A + (b * ch) * L + s; }
    case 3: { int t = m / 6, j = m - t * 6; int b = t / L, s = t - b * L;
              return A + (b * ch * L + s) * 6 + j; }
  }
}
__device__ __forceinline__ int a_stride(int mode, int L) {
  return (mode == 0) ? 1 : ((mode == 3) ? 6 * L : L);
}

__device__ __forceinline__ void gemm_store(const GemmP& p, const v8f& cc, int mbase, int n) {
  float bia = p.bias ? p.bias[n] : 0.f;
  float sc = 1.f, sh = 0.f;
  bool bn = (p.bng != nullptr);
  if (bn) { sc = p.bng[n] * rsqrtf(p.bnv[n] + 1e-5f); sh = p.bnb[n] - p.bnm[n] * sc; }
#pragma unroll
  for (int i = 0; i < 8; ++i) {
    int m = mbase + i;
    if (m >= p.M) continue;
    float v = cc[i] + bia;
    if (bn) v = v * sc + sh;
    if (p.act == 1)      v = gelu(v);
    else if (p.act == 2) v = lrelu(v);
    if (p.res) {
      if (p.resmode == 0) v += p.res[m * p.N + n];
      else { int b = m / p.L, s = m - b * p.L; v += p.res[(b * p.N + n) * p.L + s]; }
    }
    if (p.outmode == 0) p.out[m * p.N + n] = v;
    else { int b = m / p.L, s = m - b * p.L; p.out[(b * p.N + n) * p.L + s] = v; }
  }
}

__global__ __launch_bounds__(256) void gemm_wmma_kernel(GemmP p) {
  __shared__ __align__(16) _Float16 As[2][BM][ASTRIDE];
  __shared__ __align__(16) _Float16 Bs[2][BN][ASTRIDE];
  const int tid = threadIdx.x;
  const int m0 = blockIdx.x * BM;
  const int n0 = blockIdx.y * BN;
  const int wid = tid >> 5, lane = tid & 31;
  const int wm = wid & 3, wn = wid >> 2;           // 4 x 2 waves => 128 x 64
  const int l15 = lane & 15, lh = lane >> 4;

  v8f c00 = {}, c01 = {}, c10 = {}, c11 = {};

  // ---- hoisted A addressing: 4 fixed rows per thread, base + kk*stride ----
  const int ak = (tid & 7) * 4;        // 4 consecutive k
  const int ar = tid >> 3;             // row, +32 strides
  const float* aptr[4][2];
  bool mok[4];
  const int astr0 = a_stride(p.amode0, p.L);
  const int astr1 = a_stride(p.amode1, p.L);
#pragma unroll
  for (int i = 0; i < 4; ++i) {
    int m = m0 + ar + i * 32;
    mok[i] = (m < p.M);
    int mm = mok[i] ? m : 0;
    aptr[i][0] = a_base(p.A0, p.amode0, mm, p.lda0, p.ch0, p.L);
    aptr[i][1] = a_base(p.A1, p.amode1, mm, p.lda1, p.ch1, p.L);
  }
  // ---- B staging: one 16B load + one ds_store_b128 per thread ----
  const int bn_ = tid >> 2;            // 0..63 : n within tile
  const int bq  = (tid & 3) * 8;       // 8 consecutive k
  const _Float16* wrow = p.Bw + (n0 + bn_) * p.K + bq;

  float areg[4][4];
  v8h breg;

  auto load_global = [&](int k0) {
    int part = (k0 < p.ksplit) ? 0 : 1;
    int kk = k0 + ak - (part ? p.ksplit : 0);
    int st = part ? astr1 : astr0;
#pragma unroll
    for (int i = 0; i < 4; ++i) {
      if (!mok[i]) { areg[i][0] = areg[i][1] = areg[i][2] = areg[i][3] = 0.f; continue; }
      const float* ap = aptr[i][part] + kk * st;
      if (st == 1) {
        float4 v = *(const float4*)ap;
        areg[i][0] = v.x; areg[i][1] = v.y; areg[i][2] = v.z; areg[i][3] = v.w;
      } else {
        areg[i][0] = ap[0]; areg[i][1] = ap[st]; areg[i][2] = ap[2 * st]; areg[i][3] = ap[3 * st];
      }
    }
    breg = *(const v8h*)(wrow + k0);
  };
  auto store_lds = [&](int buf) {
#pragma unroll
    for (int i = 0; i < 4; ++i) {
      v4h h = { (_Float16)areg[i][0], (_Float16)areg[i][1],
                (_Float16)areg[i][2], (_Float16)areg[i][3] };
      *(v4h*)&As[buf][ar + i * 32][ak] = h;
    }
    *(v8h*)&Bs[buf][bn_][bq] = breg;
  };

  const int nc = p.K / BK;
  load_global(0);
  store_lds(0);
  __syncthreads();

  for (int c = 0; c < nc; ++c) {
    const int cur = c & 1;
    if (c + 1 < nc) load_global((c + 1) * BK);   // overlap with WMMAs below

    union F { v16h v; v8h h[2]; } a0, a1, b0, b1;
    // A 16x32 frag: lanes 0-15 M=0..15; vgpr0-3: K=lh*8+0..7, vgpr4-7: K=16+lh*8+0..7
    a0.h[0] = *(const v8h*)&As[cur][wm * 32 + l15][lh * 8];
    a0.h[1] = *(const v8h*)&As[cur][wm * 32 + l15][16 + lh * 8];
    a1.h[0] = *(const v8h*)&As[cur][wm * 32 + 16 + l15][lh * 8];
    a1.h[1] = *(const v8h*)&As[cur][wm * 32 + 16 + l15][16 + lh * 8];
    // B 32x16 frag: lane = column n; vgpr0-3: K=lh*8+0..7, vgpr4-7: K=16+lh*8+0..7
    b0.h[0] = *(const v8h*)&Bs[cur][wn * 32 + l15][lh * 8];
    b0.h[1] = *(const v8h*)&Bs[cur][wn * 32 + l15][16 + lh * 8];
    b1.h[0] = *(const v8h*)&Bs[cur][wn * 32 + 16 + l15][lh * 8];
    b1.h[1] = *(const v8h*)&Bs[cur][wn * 32 + 16 + l15][16 + lh * 8];

    c00 = __builtin_amdgcn_wmma_f32_16x16x32_f16(false, a0.v, false, b0.v, (short)0, c00, false, false);
    c01 = __builtin_amdgcn_wmma_f32_16x16x32_f16(false, a0.v, false, b1.v, (short)0, c01, false, false);
    c10 = __builtin_amdgcn_wmma_f32_16x16x32_f16(false, a1.v, false, b0.v, (short)0, c10, false, false);
    c11 = __builtin_amdgcn_wmma_f32_16x16x32_f16(false, a1.v, false, b1.v, (short)0, c11, false, false);

    if (c + 1 < nc) store_lds(1 - cur);          // waits on the global loads
    __syncthreads();
  }

  // C/D layout: vgpr i -> m = i + lh*8; n = l15
  gemm_store(p, c00, m0 + wm * 32 + lh * 8,      n0 + wn * 32 + l15);
  gemm_store(p, c01, m0 + wm * 32 + lh * 8,      n0 + wn * 32 + 16 + l15);
  gemm_store(p, c10, m0 + wm * 32 + 16 + lh * 8, n0 + wn * 32 + l15);
  gemm_store(p, c11, m0 + wm * 32 + 16 + lh * 8, n0 + wn * 32 + 16 + l15);
}

// ---------------------------------------------------------------------------
// Weight convert + transpose: dst[n*K + k] = (f16) src[k*N + n]
// ---------------------------------------------------------------------------
__global__ void wcvt_kernel(const float* __restrict__ src, _Float16* __restrict__ dst,
                            int K, int N) {
  int idx = blockIdx.x * 256 + threadIdx.x;
  if (idx >= K * N) return;
  int n = idx / K, k = idx - n * K;
  dst[idx] = (_Float16)src[k * N + n];
}

// ---------------------------------------------------------------------------
// LayerNorm over C=256. One wave32 per row, 8 elements per lane.
// mode 0: src plain row-major.  mode 1: src channel-first (b*C*L + c*L + s).
// dst always plain row-major.
// ---------------------------------------------------------------------------
__global__ void ln_kernel(const float* __restrict__ src, float* __restrict__ dst,
                          const float* __restrict__ g, const float* __restrict__ be,
                          int rows, int mode, int L) {
  int wid = threadIdx.x >> 5, lane = threadIdx.x & 31;
  int r = blockIdx.x * 8 + wid;
  if (r >= rows) return;
  int base, stride;
  if (mode == 0) { base = r * CC; stride = 1; }
  else { int b = r / L, s = r - b * L; base = b * CC * L + s; stride = L; }
  float x[8], sum = 0.f, ss = 0.f;
#pragma unroll
  for (int j = 0; j < 8; ++j) {
    int c = lane * 8 + j;
    float v = src[base + c * stride];
    x[j] = v; sum += v; ss += v * v;
  }
  sum = waveRedSum(sum); ss = waveRedSum(ss);
  float mu = sum * (1.f / CC);
  float rs = rsqrtf(ss * (1.f / CC) - mu * mu + 1e-5f);
#pragma unroll
  for (int j = 0; j < 8; ++j) {
    int c = lane * 8 + j;
    dst[r * CC + c] = (x[j] - mu) * rs * g[c] + be[c];
  }
}

// ---------------------------------------------------------------------------
// priors gather: P0[b,i,c] = domain_prior[mod_idx[b], i, c]
// ---------------------------------------------------------------------------
__global__ void gather_prior_kernel(const float* __restrict__ dp, const int* __restrict__ midx,
                                    float* __restrict__ p0) {
  int idx = blockIdx.x * 256 + threadIdx.x;       // B*LP*CC = 25600
  int b = idx / (LP * CC);
  p0[idx] = dp[midx[b] * (LP * CC) + (idx - b * (LP * CC))];
}

// ---------------------------------------------------------------------------
// Attention, few queries vs many keys (ca1): one block per (b, head, query).
// Q: (B,LP,256) head-packed.  KV: (B,LL,512) = [k(256) | v(256)].
// ---------------------------------------------------------------------------
__global__ void attn_fewq_kernel(const float* __restrict__ Q, const float* __restrict__ KV,
                                 float* __restrict__ O) {
  __shared__ float qs[32];
  __shared__ float prob[LL];
  __shared__ float red[256];
  const float scale = 0.17677669529663687f;       // 32^-0.5
  int tid = threadIdx.x, bid = blockIdx.x;
  int i = bid % LP; int hd = (bid / LP) % NH; int b = bid / (LP * NH);
  if (tid < 32) qs[tid] = Q[(b * LP + i) * CC + hd * DH + tid];
  __syncthreads();
  float lmax = -1e30f;
  for (int l = tid; l < LL; l += 256) {
    const float* kp = KV + (b * LL + l) * 512 + hd * DH;
    float acc = 0.f;
#pragma unroll
    for (int u = 0; u < DH; ++u) acc += qs[u] * kp[u];
    acc *= scale;
    prob[l] = acc;
    lmax = fmaxf(lmax, acc);
  }
  red[tid] = lmax; __syncthreads();
  for (int s = 128; s > 0; s >>= 1) { if (tid < s) red[tid] = fmaxf(red[tid], red[tid + s]); __syncthreads(); }
  float mx = red[0]; __syncthreads();
  float lsum = 0.f;
  for (int l = tid; l < LL; l += 256) { float e = __expf(prob[l] - mx); prob[l] = e; lsum += e; }
  red[tid] = lsum; __syncthreads();
  for (int s = 128; s > 0; s >>= 1) { if (tid < s) red[tid] += red[tid + s]; __syncthreads(); }
  float inv = 1.f / red[0]; __syncthreads();
  int wid = tid >> 5, lane = tid & 31;
  float acc = 0.f;
  const int chunk = LL / 8;
  for (int l = wid * chunk; l < (wid + 1) * chunk; ++l)
    acc += prob[l] * KV[(b * LL + l) * 512 + 256 + hd * DH + lane];
  red[wid * 32 + lane] = acc;
  __syncthreads();
  if (wid == 0) {
    float t = 0.f;
#pragma unroll
    for (int w = 0; w < 8; ++w) t += red[w * 32 + lane];
    O[(b * LP + i) * CC + hd * DH + lane] = t * inv;
  }
}

// ---------------------------------------------------------------------------
// Attention, many queries vs 50 keys (ca2): one wave per (b, head, query).
// ---------------------------------------------------------------------------
__global__ void attn_manyq_kernel(const float* __restrict__ Q, const float* __restrict__ KV,
                                  float* __restrict__ O) {
  __shared__ float lg[8][64];
  const float scale = 0.17677669529663687f;
  int tid = threadIdx.x, wid = tid >> 5, u = tid & 31;
  int r = blockIdx.x * 8 + wid;                   // over B*NH*LL = 65536
  int l = r % LL; int hd = (r / LL) % NH; int b = r / (LL * NH);
  float qv = Q[(b * LL + l) * CC + hd * DH + u];
  for (int j = 0; j < LP; ++j) {
    float d = qv * KV[(b * LP + j) * 512 + hd * DH + u];
    d = waveRedSum(d);
    if (u == 0) lg[wid][j] = d * scale;
  }
  __syncthreads();
  float v0 = (u < LP) ? lg[wid][u] : -1e30f;
  float v1 = (u + 32 < LP) ? lg[wid][u + 32] : -1e30f;
  float mx = waveRedMax(fmaxf(v0, v1));
  float e0 = (u < LP) ? __expf(v0 - mx) : 0.f;
  float e1 = (u + 32 < LP) ? __expf(v1 - mx) : 0.f;
  float sm = waveRedSum(e0 + e1);
  if (u < LP) lg[wid][u] = e0;
  if (u + 32 < LP) lg[wid][u + 32] = e1;
  __syncthreads();
  float inv = 1.f / sm;
  float acc = 0.f;
  for (int j = 0; j < LP; ++j)
    acc += lg[wid][j] * KV[(b * LP + j) * 512 + 256 + hd * DH + u];
  O[(b * LL + l) * CC + hd * DH + u] = acc * inv;
}

// ---------------------------------------------------------------------------
// TPFM edge build + 'ea' 6->1 reduce (+scalar BN + lrelu).
// EDGE[((b*C+c)*L+s)*6+j] ; AGG channel-first [(b*C+c)*L+s].
// ---------------------------------------------------------------------------
__global__ void edge_ea_kernel(const float* __restrict__ X, const float* __restrict__ topo,
                               const float* __restrict__ eaW, const float* __restrict__ eab,
                               const float* __restrict__ g, const float* __restrict__ be,
                               const float* __restrict__ mm, const float* __restrict__ vv,
                               float* __restrict__ EDGE, float* __restrict__ AGG) {
  int idx = blockIdx.x * 256 + threadIdx.x;       // B*C*L
  int s = idx & (LL - 1);
  int base = idx - s;
  int d = s >> 8, hh = (s >> 4) & 15, ww = s & 15;
  float xc = X[idx];
  int n0 = (((d + 15) & 15) << 8) | (hh << 4) | ww;   // roll +1 on D -> x[d-1]
  int n1 = (((d + 1)  & 15) << 8) | (hh << 4) | ww;
  int n2 = (d << 8) | (((hh + 15) & 15) << 4) | ww;
  int n3 = (d << 8) | (((hh + 1)  & 15) << 4) | ww;
  int n4 = (d << 8) | (hh << 4) | ((ww + 15) & 15);
  int n5 = (d << 8) | (hh << 4) | ((ww + 1)  & 15);
  float e[6];
  e[0] = X[base + n0] * xc + topo[s * 6 + 0];
  e[1] = X[base + n1] * xc + topo[s * 6 + 1];
  e[2] = X[base + n2] * xc + topo[s * 6 + 2];
  e[3] = X[base + n3] * xc + topo[s * 6 + 3];
  e[4] = X[base + n4] * xc + topo[s * 6 + 4];
  e[5] = X[base + n5] * xc + topo[s * 6 + 5];
  float* ep = EDGE + idx * 6;
  float a = eab[0];
#pragma unroll
  for (int j = 0; j < 6; ++j) { ep[j] = e[j]; a += e[j] * eaW[j]; }
  float sc = g[0] * rsqrtf(vv[0] + 1e-5f);
  a = (a - mm[0]) * sc + be[0];
  AGG[idx] = lrelu(a);
}

// ---------------------------------------------------------------------------
// 'er' 6->1 reduce over UE (+scalar BN + lrelu), fused with uf = uv * uec.
// ---------------------------------------------------------------------------
__global__ void er_uf_kernel(const float* __restrict__ UE, const float* __restrict__ UV,
                             const float* __restrict__ erW, const float* __restrict__ erb,
                             const float* __restrict__ g, const float* __restrict__ be,
                             const float* __restrict__ mm, const float* __restrict__ vv,
                             float* __restrict__ UF) {
  int idx = blockIdx.x * 256 + threadIdx.x;       // B*L*128
  int m = idx >> 7, c = idx & 127;
  float a = erb[0];
#pragma unroll
  for (int j = 0; j < 6; ++j) a += UE[(m * 6 + j) * 128 + c] * erW[j];
  float sc = g[0] * rsqrtf(vv[0] + 1e-5f);
  a = (a - mm[0]) * sc + be[0];
  UF[idx] = UV[idx] * lrelu(a);
}

// ---------------------------------------------------------------------------
// Host side
// ---------------------------------------------------------------------------
static inline GemmP gp_base(const float* A, const _Float16* Bw, float* out, int M, int N, int K) {
  GemmP p{};
  p.A0 = A; p.A1 = A; p.Bw = Bw; p.out = out;
  p.M = M; p.N = N; p.K = K; p.ksplit = K;
  p.amode0 = 0; p.amode1 = 0; p.lda0 = K; p.lda1 = K; p.ch0 = K; p.ch1 = 0;
  p.L = LL; p.act = 0; p.resmode = 0; p.outmode = 0;
  p.bias = nullptr; p.bng = nullptr; p.bnb = nullptr; p.bnm = nullptr; p.bnv = nullptr;
  p.res = nullptr;
  return p;
}
static inline void launch_gemm(const GemmP& p, hipStream_t st) {
  dim3 g((unsigned)((p.M + BM - 1) / BM), (unsigned)(p.N / BN));
  gemm_wmma_kernel<<<g, dim3(256), 0, st>>>(p);
}
static inline void wcvt(const float* s, _Float16* d, int K, int N, hipStream_t st) {
  int tot = K * N;
  wcvt_kernel<<<(tot + 255) / 256, 256, 0, st>>>(s, d, K, N);
}

extern "C" void kernel_launch(void* const* d_in, const int* in_sizes, int n_in,
                              void* d_out, int out_size, void* d_ws, size_t ws_size,
                              hipStream_t stream) {
  (void)in_sizes; (void)n_in; (void)out_size;
  const float* const* in = (const float* const*)d_in;

  // ---- flattened input leaves (depth-first dict/tuple order) ----
  const float *X = in[0], *DPRIOR = in[1];
  const float *LP1g = in[2],  *LP1b = in[3],  *LF1g = in[4],  *LF1b = in[5];
  const float *Wq1 = in[6], *Wkv1 = in[7], *Wo1 = in[8], *bo1 = in[9];
  const float *LPFg = in[10], *LPFb = in[11];
  const float *W1p = in[12], *b1p = in[13], *W2p = in[14], *b2p = in[15];
  const float *LF2g = in[16], *LF2b = in[17], *LP2g = in[18], *LP2b = in[19];
  const float *Wq2 = in[20], *Wkv2 = in[21], *Wo2 = in[22], *bo2 = in[23];
  const float *LFFg = in[24], *LFFb = in[25];
  const float *W1f = in[26], *b1f = in[27], *W2f = in[28], *b2f = in[29];
  const float *TOPO = in[30];
  const float *eaW = in[31], *eab = in[32], *eag = in[33], *eabb = in[34], *eam = in[35], *eav = in[36];
  const float *vuW = in[37], *vub = in[38], *vug = in[39], *vubb = in[40], *vum = in[41], *vuv = in[42];
  const float *euW = in[43], *eub = in[44], *eug = in[45], *eubb = in[46], *eum = in[47], *euv = in[48];
  const float *erW = in[49], *erb = in[50], *erg = in[51], *erbb = in[52], *erm = in[53], *erv = in[54];
  const float *faW = in[55], *fag = in[56], *fabb = in[57], *fam = in[58], *fav = in[59];
  const float *convW = in[60], *convB = in[61];
  const int* MODIDX = (const int*)d_in[62];

  // ---- workspace layout: float region, then f16 transposed-weight region ----
  const size_t OF_P0   = 0;
  const size_t OF_P1   = OF_P0   + 25600;
  const size_t OF_LNP  = OF_P1   + 25600;
  const size_t OF_Q1   = OF_LNP  + 25600;
  const size_t OF_O1   = OF_Q1   + 25600;
  const size_t OF_HP   = OF_O1   + 25600;
  const size_t OF_KV1  = OF_HP   + 102400;
  const size_t OF_LNX  = OF_KV1  + 4194304;
  const size_t OF_Q2   = OF_LNX  + 2097152;
  const size_t OF_KV2  = OF_Q2   + 2097152;
  const size_t OF_O2   = OF_KV2  + 51200;
  const size_t OF_X1   = OF_O2   + 2097152;
  const size_t OF_HF   = OF_X1   + 2097152;
  const size_t OF_XDP  = OF_HF   + 8388608;
  const size_t OF_EDGE = OF_XDP  + 2097152;
  const size_t OF_AGG  = OF_EDGE + 12582912;
  const size_t OF_UV   = OF_AGG  + 2097152;
  const size_t OF_UE   = OF_UV   + 1048576;
  const size_t OF_UF   = OF_UE   + 6291456;
  const size_t OF_XTP  = OF_UF   + 1048576;
  const size_t TOTAL_F = OF_XTP  + 2097152;
  const size_t W16_TOT = 1933312;                 // f16 elements
  if (ws_size < TOTAL_F * sizeof(float) + W16_TOT * sizeof(_Float16)) return;

  float* ws = (float*)d_ws;
  float* P0 = ws + OF_P0;   float* P1 = ws + OF_P1;  float* LNPb = ws + OF_LNP;
  float* Q1 = ws + OF_Q1;   float* O1 = ws + OF_O1;  float* HP = ws + OF_HP;
  float* KV1 = ws + OF_KV1; float* LNX = ws + OF_LNX; float* Q2 = ws + OF_Q2;
  float* KV2 = ws + OF_KV2; float* O2 = ws + OF_O2;  float* X1 = ws + OF_X1;
  float* HF = ws + OF_HF;   float* XDP = ws + OF_XDP; float* EDGE = ws + OF_EDGE;
  float* AGG = ws + OF_AGG; float* UV = ws + OF_UV;  float* UE = ws + OF_UE;
  float* UF = ws + OF_UF;   float* XTP = ws + OF_XTP;

  _Float16* w16 = (_Float16*)(ws + TOTAL_F);
  _Float16* Wq1t  = w16;            // 65536   (K=256,  N=256)
  _Float16* Wkv1t = w16 + 65536;    // 131072  (K=256,  N=512)
  _Float16* Wo1t  = w16 + 196608;   // 65536
  _Float16* W1pt  = w16 + 262144;   // 262144  (K=256,  N=1024)
  _Float16* W2pt  = w16 + 524288;   // 262144  (K=1024, N=256)
  _Float16* Wq2t  = w16 + 786432;   // 65536
  _Float16* Wkv2t = w16 + 851968;   // 131072
  _Float16* Wo2t  = w16 + 983040;   // 65536
  _Float16* W1ft  = w16 + 1048576;  // 262144
  _Float16* W2ft  = w16 + 1310720;  // 262144
  _Float16* vuWt  = w16 + 1572864;  // 65536   (K=512,  N=128)
  _Float16* euWt  = w16 + 1638400;  // 65536   (K=512,  N=128)
  _Float16* faWt  = w16 + 1703936;  // 98304   (K=384,  N=256)
  _Float16* convWt= w16 + 1802240;  // 131072  (K=512,  N=256)

  float* OUT    = (float*)d_out;             // (B, 256, D, H, W) channel-first
  float* PRIORS = (float*)d_out + 2097152;   // (B, 50, 256)

  const int rowsX = BB * LL;                 // 8192
  const int rowsP = BB * LP;                 // 100

  // ---- one-time (per launch) weight convert/transpose to f16 [N][K] ----
  wcvt(Wq1, Wq1t, 256, 256, stream);   wcvt(Wkv1, Wkv1t, 256, 512, stream);
  wcvt(Wo1, Wo1t, 256, 256, stream);   wcvt(W1p, W1pt, 256, 1024, stream);
  wcvt(W2p, W2pt, 1024, 256, stream);  wcvt(Wq2, Wq2t, 256, 256, stream);
  wcvt(Wkv2, Wkv2t, 256, 512, stream); wcvt(Wo2, Wo2t, 256, 256, stream);
  wcvt(W1f, W1ft, 256, 1024, stream);  wcvt(W2f, W2ft, 1024, 256, stream);
  wcvt(vuW, vuWt, 512, 128, stream);   wcvt(euW, euWt, 512, 128, stream);
  wcvt(faW, faWt, 384, 256, stream);   wcvt(convW, convWt, 512, 256, stream);

  // ================= DPAM =================
  gather_prior_kernel<<<100, 256, 0, stream>>>(DPRIOR, MODIDX, P0);
  ln_kernel<<<(rowsP + 7) / 8, 256, 0, stream>>>(P0, LNPb, LP1g, LP1b, rowsP, 0, LL);
  ln_kernel<<<rowsX / 8, 256, 0, stream>>>(X, LNX, LF1g, LF1b, rowsX, 1, LL);

  { GemmP p = gp_base(LNPb, Wq1t, Q1, rowsP, CC, CC); launch_gemm(p, stream); }
  { GemmP p = gp_base(LNX, Wkv1t, KV1, rowsX, 512, CC); launch_gemm(p, stream); }
  attn_fewq_kernel<<<BB * NH * LP, 256, 0, stream>>>(Q1, KV1, O1);
  { GemmP p = gp_base(O1, Wo1t, P1, rowsP, CC, CC);
    p.bias = bo1; p.res = P0; p.resmode = 0; launch_gemm(p, stream); }

  ln_kernel<<<(rowsP + 7) / 8, 256, 0, stream>>>(P1, LNPb, LPFg, LPFb, rowsP, 0, LL);
  { GemmP p = gp_base(LNPb, W1pt, HP, rowsP, 1024, CC); p.bias = b1p; p.act = 1; launch_gemm(p, stream); }
  { GemmP p = gp_base(HP, W2pt, PRIORS, rowsP, CC, 1024); p.bias = b2p; p.res = P1; launch_gemm(p, stream); }

  ln_kernel<<<rowsX / 8, 256, 0, stream>>>(X, LNX, LF2g, LF2b, rowsX, 1, LL);
  { GemmP p = gp_base(LNX, Wq2t, Q2, rowsX, CC, CC); launch_gemm(p, stream); }
  ln_kernel<<<(rowsP + 7) / 8, 256, 0, stream>>>(PRIORS, LNPb, LP2g, LP2b, rowsP, 0, LL);
  { GemmP p = gp_base(LNPb, Wkv2t, KV2, rowsP, 512, CC); launch_gemm(p, stream); }
  attn_manyq_kernel<<<BB * NH * LL / 8, 256, 0, stream>>>(Q2, KV2, O2);
  { GemmP p = gp_base(O2, Wo2t, X1, rowsX, CC, CC);
    p.bias = bo2; p.res = X; p.resmode = 1; launch_gemm(p, stream); }  // +xf (channel-first x)

  ln_kernel<<<rowsX / 8, 256, 0, stream>>>(X1, LNX, LFFg, LFFb, rowsX, 0, LL);
  { GemmP p = gp_base(LNX, W1ft, HF, rowsX, 1024, CC); p.bias = b1f; p.act = 1; launch_gemm(p, stream); }
  { GemmP p = gp_base(HF, W2ft, XDP, rowsX, CC, 1024); p.bias = b2f; p.res = X1; launch_gemm(p, stream); }

  // ================= TPFM =================
  edge_ea_kernel<<<BB * CC * LL / 256, 256, 0, stream>>>(X, TOPO, eaW, eab, eag, eabb, eam, eav, EDGE, AGG);

  // uv = lrelu(bn(catv @ vuW + vub)); catv = [x | agg] channel-last view
  { GemmP p = gp_base(X, vuWt, UV, rowsX, 128, 512);
    p.A0 = X; p.A1 = AGG; p.ksplit = 256; p.amode0 = 1; p.amode1 = 1; p.ch0 = CC; p.ch1 = CC;
    p.bias = vub; p.bng = vug; p.bnb = vubb; p.bnm = vum; p.bnv = vuv; p.act = 2;
    launch_gemm(p, stream); }

  // ue = lrelu(bn(cate @ euW + eub)); cate = [x bcast over j | edge] view; M = B*L*6
  { GemmP p = gp_base(X, euWt, UE, rowsX * 6, 128, 512);
    p.A0 = X; p.A1 = EDGE; p.ksplit = 256; p.amode0 = 2; p.amode1 = 3; p.ch0 = CC; p.ch1 = CC;
    p.bias = eub; p.bng = eug; p.bnb = eubb; p.bnm = eum; p.bnv = euv; p.act = 2;
    launch_gemm(p, stream); }

  er_uf_kernel<<<BB * LL * 128 / 256, 256, 0, stream>>>(UE, UV, erW, erb, erg, erbb, erm, erv, UF);

  // x_tp = lrelu(bn_ch([x | uf] @ faW)); stored channel-last (B*L, 256)
  { GemmP p = gp_base(X, faWt, XTP, rowsX, CC, 384);
    p.A0 = X; p.A1 = UF; p.ksplit = 256; p.amode0 = 1; p.amode1 = 0; p.ch0 = CC; p.lda1 = 128;
    p.bng = fag; p.bnb = fabb; p.bnm = fam; p.bnv = fav; p.act = 2;
    launch_gemm(p, stream); }

  // ================= final conv =================
  // out = [x_dp | x_tp] @ convW + convB, stored channel-first
  { GemmP p = gp_base(XDP, convWt, OUT, rowsX, CC, 512);
    p.A0 = XDP; p.A1 = XTP; p.ksplit = 256; p.amode0 = 0; p.amode1 = 0; p.lda0 = 256; p.lda1 = 256;
    p.bias = convB; p.outmode = 1;
    launch_gemm(p, stream); }
}
